// NeuReparam_31482110280013
// MI455X (gfx1250) — compile-verified, matrix operand order checked by reading, not run on hardware
//
#include <hip/hip_runtime.h>
#include <hip/hip_bf16.h>

typedef __attribute__((ext_vector_type(16))) _Float16 v16h;
typedef __attribute__((ext_vector_type(8)))  float    v8f;

#define BATCH 524288
#define WAVES 4
#define BLOCK (WAVES * 32)

#define TILE_STRIDE 72                 // halfs per row (64 cols + pad, keeps 16B align)
#define TILE_HALFS  (16 * TILE_STRIDE)
#define TILE_BYTES  (TILE_HALFS * 2)   // 2304 B

#define FRAG_HALFS 512                 // 32 lanes * 16 halfs
#define NFRAG_L0   4
#define NFRAG_HID  24                  // 3 layers * 2 kc * 4 nt
#define NFRAG_OUT  2
#define NFRAGS     (NFRAG_L0 + NFRAG_HID + NFRAG_OUT)   // 30
#define WFRAG_BYTES (NFRAGS * FRAG_HALFS * 2)           // 30720 B
#define SMEM_BYTES  (WFRAG_BYTES + WAVES * 3 * TILE_BYTES) // 58368 B

// Half index -> K index inside a 16x32 f16 A/B WMMA fragment (ISA 7.12.2):
// lanes 0-15: VGPR0-3 hold K=0..7, VGPR4-7 hold K=16..23; lanes 16-31: +8.
__device__ __forceinline__ int khalf(int t, int hi) {
    return (t < 8 ? t : t + 8) + (hi ? 8 : 0);
}

// Single-instruction v_rcp_f32 (avoid IEEE div_scale/div_fmas expansion).
__device__ __forceinline__ float fast_rcp(float x) {
    return __builtin_amdgcn_rcpf(x);
}

__device__ __forceinline__ float fast_sigmoid(float x) {
    return fast_rcp(1.0f + __expf(-x));
}

__device__ __forceinline__ v16h load_frag(const _Float16* fragbase, int lane) {
    union { uint4 q[2]; v16h h; } u;
    const uint4* p = (const uint4*)(fragbase + lane * 16);
    u.q[0] = p[0];
    u.q[1] = p[1];
    return u.h;
}

// Load an A fragment (16x32 slice starting at K = kc*32) from a row-major
// f16 tile in LDS. Thanks to the K layout, this is 2x contiguous b128 loads.
__device__ __forceinline__ v16h load_a(const _Float16* tile, int lane, int kc) {
    int m = lane & 15;
    int kbase = kc * 32 + ((lane >= 16) ? 8 : 0);
    const _Float16* row = tile + m * TILE_STRIDE + kbase;
    union { uint4 q[2]; v16h h; } u;
    u.q[0] = *(const uint4*)(row);
    u.q[1] = *(const uint4*)(row + 16);
    return u.h;
}

// Scatter a 16x16 f32 D fragment (C/D layout, ISA 7.12.2) into a row-major
// f16 LDS tile at column offset nt*16.
__device__ __forceinline__ void store_tile(_Float16* tile, int lane, int nt,
                                           const float* v) {
    int n  = nt * 16 + (lane & 15);
    int mo = (lane >= 16) ? 8 : 0;
#pragma unroll
    for (int r = 0; r < 8; ++r)
        tile[(r + mo) * TILE_STRIDE + n] = (_Float16)v[r];
}

__device__ __forceinline__ float feat(int k, float wo0, float wo1,
                                      float z0, float z1) {
    if (k >= 20) return 0.0f;
    if (k == 18) return z0;
    if (k == 19) return z1;
    if (k == 0)  return wo0;
    if (k == 1)  return wo1;
    int i = (k - 2) >> 2;
    int r = (k - 2) & 3;
    float x = (r & 1) ? wo1 : wo0;
    float f = x * (float)(1 << i);
    return (r < 2) ? __sinf(f) : __cosf(f);
}

__device__ __forceinline__ v8f wmma(v16h a, v16h b, v8f c) {
    return __builtin_amdgcn_wmma_f32_16x16x32_f16(false, a, false, b,
                                                  (short)0, c, false, false);
}

__global__ __launch_bounds__(BLOCK)
void NeuReparam_31482110280013_kernel(const float* __restrict__ wo,
                                      const float* __restrict__ z,
                                      const float* __restrict__ W_in,
                                      const float* __restrict__ b_in,
                                      const float* __restrict__ W_hid,
                                      const float* __restrict__ b_hid,
                                      const float* __restrict__ W_out,
                                      const float* __restrict__ b_out,
                                      float* __restrict__ out) {
    __shared__ __align__(16) unsigned char smem[SMEM_BYTES];
    _Float16* frag_lds = (_Float16*)smem;

    const int tid = threadIdx.x;

    // ---- Pack all weights into LDS in WMMA B-fragment order (f16) ----
    for (int idx = tid; idx < NFRAGS * FRAG_HALFS; idx += BLOCK) {
        int f = idx >> 9;
        int s = idx & 511;
        int lane = s >> 4, t = s & 15;
        int hi = (lane >= 16);
        int n16 = lane & 15;
        float val = 0.0f;
        if (f < NFRAG_L0) {                       // layer 0: K=20 padded to 32
            int k = khalf(t, hi);
            int n = f * 16 + n16;
            if (k < 20) val = W_in[k * 64 + n];
        } else if (f < NFRAG_L0 + NFRAG_HID) {    // hidden layers: K=64
            int i = f - NFRAG_L0;
            int l = i >> 3, r = i & 7;
            int kc = r >> 2, nt = r & 3;
            int k = kc * 32 + khalf(t, hi);
            int n = nt * 16 + n16;
            val = W_hid[(l * 64 + k) * 64 + n];
        } else {                                  // output layer: N=3 padded to 16
            int kc = f - (NFRAG_L0 + NFRAG_HID);
            int k = kc * 32 + khalf(t, hi);
            if (n16 < 3) val = W_out[k * 3 + n16];
        }
        frag_lds[idx] = (_Float16)val;
    }
    __syncthreads();

    const int wave = tid >> 5;
    const int lane = tid & 31;
    const int m    = lane & 15;
    const int hi   = (lane >= 16);
    const int m0   = (blockIdx.x * WAVES + wave) * 16;
    const int row  = m0 + m;

    _Float16* tiles = (_Float16*)(smem + WFRAG_BYTES + wave * 3 * TILE_BYTES);
    _Float16* tH = tiles;
    _Float16* tU = tiles + TILE_HALFS;
    _Float16* tV = tiles + 2 * TILE_HALFS;

    const float wo0 = wo[row * 2 + 0], wo1 = wo[row * 2 + 1];
    const float z0  = z[row * 2 + 0],  z1  = z[row * 2 + 1];

    // ---- Layer 0 A fragment built directly in registers (posenc) ----
    v16h a0;
#pragma unroll
    for (int t = 0; t < 16; ++t) {
        int k = khalf(t, hi);
        a0[t] = (_Float16)feat(k, wo0, wo1, z0, z1);
    }

    // ---- Layer 0: primal GEMM; tangents injected analytically ----
#pragma unroll
    for (int nt = 0; nt < 4; ++nt) {
        int n = nt * 16 + m;
        float bias = b_in[n];
        v8f c;
#pragma unroll
        for (int j = 0; j < 8; ++j) c[j] = bias;
        v16h bf = load_frag(frag_lds + nt * FRAG_HALFS, lane);
        c = wmma(a0, bf, c);

        float w18 = W_in[18 * 64 + n];
        float w19 = W_in[19 * 64 + n];
        float hv[8], uv[8], vv[8];
#pragma unroll
        for (int j = 0; j < 8; ++j) {
            float x  = c[j];
            float sg = fast_sigmoid(x);
            float sp = __builtin_fmaf(x * sg, (1.0f - sg), sg);   // d silu
            hv[j] = x * sg;
            uv[j] = sp * w18;   // tangent0 pre-act = W_in row 18
            vv[j] = sp * w19;   // tangent1 pre-act = W_in row 19
        }
        store_tile(tH, lane, nt, hv);
        store_tile(tU, lane, nt, uv);
        store_tile(tV, lane, nt, vv);
    }

    // ---- Hidden layers: 3 streams share each B fragment ----
    for (int l = 0; l < 3; ++l) {
        v16h aH0 = load_a(tH, lane, 0), aH1 = load_a(tH, lane, 1);
        v16h aU0 = load_a(tU, lane, 0), aU1 = load_a(tU, lane, 1);
        v16h aV0 = load_a(tV, lane, 0), aV1 = load_a(tV, lane, 1);
#pragma unroll
        for (int nt = 0; nt < 4; ++nt) {
            float bias = b_hid[l * 64 + nt * 16 + m];
            v8f cP, cU, cV;
#pragma unroll
            for (int j = 0; j < 8; ++j) { cP[j] = bias; cU[j] = 0.0f; cV[j] = 0.0f; }
#pragma unroll
            for (int kc = 0; kc < 2; ++kc) {
                int f = NFRAG_L0 + l * 8 + kc * 4 + nt;
                v16h bf = load_frag(frag_lds + f * FRAG_HALFS, lane);
                cP = wmma(kc ? aH1 : aH0, bf, cP);
                cU = wmma(kc ? aU1 : aU0, bf, cU);
                cV = wmma(kc ? aV1 : aV0, bf, cV);
            }
            float hv[8], uv[8], vv[8];
#pragma unroll
            for (int j = 0; j < 8; ++j) {
                float x  = cP[j];
                float sg = fast_sigmoid(x);
                float sp = __builtin_fmaf(x * sg, (1.0f - sg), sg);
                hv[j] = x * sg;
                uv[j] = sp * cU[j];
                vv[j] = sp * cV[j];
            }
            store_tile(tH, lane, nt, hv);
            store_tile(tU, lane, nt, uv);
            store_tile(tV, lane, nt, vv);
        }
    }

    // ---- Output layer (N padded 3 -> 16, only nt = 0) ----
    {
        v16h aH0 = load_a(tH, lane, 0), aH1 = load_a(tH, lane, 1);
        v16h aU0 = load_a(tU, lane, 0), aU1 = load_a(tU, lane, 1);
        v16h aV0 = load_a(tV, lane, 0), aV1 = load_a(tV, lane, 1);
        float bias = (m < 3) ? b_out[m] : 0.0f;
        v8f cP, cU, cV;
#pragma unroll
        for (int j = 0; j < 8; ++j) { cP[j] = bias; cU[j] = 0.0f; cV[j] = 0.0f; }
#pragma unroll
        for (int kc = 0; kc < 2; ++kc) {
            int f = NFRAG_L0 + NFRAG_HID + kc;
            v16h bf = load_frag(frag_lds + f * FRAG_HALFS, lane);
            cP = wmma(kc ? aH1 : aH0, bf, cP);
            cU = wmma(kc ? aU1 : aU0, bf, cU);
            cV = wmma(kc ? aV1 : aV0, bf, cV);
        }
        // Stage f32 D tiles into (now free) LDS tiles for the cross-lane epilogue.
        float* sH = (float*)tH;
        float* sU = (float*)tU;
        float* sV = (float*)tV;
        int mo = hi ? 8 : 0;
#pragma unroll
        for (int r = 0; r < 8; ++r) {
            sH[(r + mo) * 16 + m] = cP[r];
            sU[(r + mo) * 16 + m] = cU[r];
            sV[(r + mo) * 16 + m] = cV[r];
        }

        // ---- Epilogue: softplus, normalize, 2x2 det (lanes 0..15, one row each) ----
        if (lane < 16) {
            int orow = m0 + lane;
            float p0 = sH[lane * 16 + 0], p1 = sH[lane * 16 + 1], p2 = sH[lane * 16 + 2];
            float u0 = sU[lane * 16 + 0], u1 = sU[lane * 16 + 1], u2 = sU[lane * 16 + 2];
            float v0 = sV[lane * 16 + 0], v1 = sV[lane * 16 + 1], v2 = sV[lane * 16 + 2];

            float sg  = fast_sigmoid(p2);
            float w2  = (p2 > 20.0f) ? p2 : __logf(1.0f + __expf(p2));   // softplus
            float uu2 = sg * u2;
            float vv2 = sg * v2;

            float nsq = p0 * p0 + p1 * p1 + w2 * w2;
            float nrm = sqrtf(nsq);
            float nc  = fmaxf(nrm, 1e-12f);
            float inv = fast_rcp(nc);

            float wi0 = p0 * inv, wi1 = p1 * inv;

            float du = p0 * u0 + p1 * u1 + w2 * uu2;
            float dv = p0 * v0 + p1 * v1 + w2 * vv2;
            // 1/(nrm*nc*nc): nrm>eps always in practice; guard anyway.
            float g  = (nrm > 1e-12f) ? fast_rcp(nrm * nc * nc) : 0.0f;
            float gu = du * g;
            float gv = dv * g;

            float d00 = u0 * inv - p0 * gu;
            float d01 = u1 * inv - p1 * gu;
            float d10 = v0 * inv - p0 * gv;
            float d11 = v1 * inv - p1 * gv;
            float det = d00 * d11 - d10 * d01;

            out[orow * 3 + 0] = wi0;
            out[orow * 3 + 1] = wi1;
            out[orow * 3 + 2] = det;
        }
    }
}

extern "C" void kernel_launch(void* const* d_in, const int* in_sizes, int n_in,
                              void* d_out, int out_size, void* d_ws, size_t ws_size,
                              hipStream_t stream) {
    (void)in_sizes; (void)n_in; (void)out_size; (void)d_ws; (void)ws_size;
    const float* wo    = (const float*)d_in[0];
    const float* z     = (const float*)d_in[1];
    const float* W_in  = (const float*)d_in[2];
    const float* b_in  = (const float*)d_in[3];
    const float* W_hid = (const float*)d_in[4];
    const float* b_hid = (const float*)d_in[5];
    const float* W_out = (const float*)d_in[6];
    const float* b_out = (const float*)d_in[7];
    float* out = (float*)d_out;

    const int rows_per_block = WAVES * 16;            // 64
    const int grid = BATCH / rows_per_block;          // 8192
    NeuReparam_31482110280013_kernel<<<grid, BLOCK, 0, stream>>>(
        wo, z, W_in, b_in, W_hid, b_hid, W_out, b_out, out);
}
